// CausalSelfAttention_6777458393739
// MI455X (gfx1250) — matrix operand order repaired and checked by reading, not compiled, not run on hardware
//
#include <hip/hip_runtime.h>
#include <hip/hip_bf16.h>

#define S_LEN   2048
#define DM      768
#define N_HEAD  12
#define HD      64

typedef __attribute__((ext_vector_type(16))) __bf16 v16bf;
typedef __attribute__((ext_vector_type(8)))  float  v8f;

union Frag { unsigned int u[8]; v16bf v; };
union Acc  { float f[8]; v8f v; };

// A-matrix 16x32 bf16 fragment K offset (ISA 7.12.2): reg j, lane-half hi
__device__ __forceinline__ int ka_off(int j, int hi) {
  return ((j < 4) ? (2 * j) : (2 * j + 8)) + 8 * hi;
}
// B-matrix 32x16 bf16 fragment K offset: same K order as A (reg j, lane-half hi)
__device__ __forceinline__ int kb_off(int j, int hi) {
  return ((j < 4) ? (2 * j) : (2 * j + 8)) + 8 * hi;
}

__device__ __forceinline__ unsigned short f2bf(float f) {
  unsigned int u = __float_as_uint(f);
  u += 0x7fffu + ((u >> 16) & 1u);            // round-to-nearest-even
  return (unsigned short)(u >> 16);
}

// ---------------------------------------------------------------------------
// Kernel 1: out = x @ W^T (W selected by blockIdx.z), fused RoPE for Q/K,
// bf16 result stored head-major [b*H+h][s][d].
// ---------------------------------------------------------------------------
__global__ __launch_bounds__(256)
void qkv_rope_kernel(const float* __restrict__ x,
                     const float* __restrict__ Wq,
                     const float* __restrict__ Wk,
                     const float* __restrict__ Wv,
                     unsigned short* __restrict__ qo,
                     unsigned short* __restrict__ ko,
                     unsigned short* __restrict__ vo) {
  __shared__ unsigned short A_lds[128 * 34];
  __shared__ unsigned short B_lds[64 * 34];

  const int tid = threadIdx.x;
  const int wave = tid >> 5, lane = tid & 31, l16 = lane & 15, hi = lane >> 4;
  const int m0 = blockIdx.x * 128;
  const int n0 = blockIdx.y * 64;
  const int z  = blockIdx.z;
  const float* W = (z == 0) ? Wq : (z == 1) ? Wk : Wv;
  unsigned short* outp = (z == 0) ? qo : (z == 1) ? ko : vo;

  Acc acc[4];
#pragma unroll
  for (int t = 0; t < 4; ++t)
#pragma unroll
    for (int r = 0; r < 8; ++r) acc[t].f[r] = 0.0f;

  for (int k0 = 0; k0 < DM; k0 += 32) {
    __syncthreads();
    // Stage A tile 128x32 (f32 -> bf16)
#pragma unroll
    for (int i = 0; i < 8; ++i) {
      int p = tid + i * 256;
      int row = p >> 4, col = (p & 15) * 2;
      float2 f = *(const float2*)&x[(size_t)(m0 + row) * DM + k0 + col];
      unsigned int pk = (unsigned int)f2bf(f.x) | ((unsigned int)f2bf(f.y) << 16);
      *(unsigned int*)&A_lds[row * 34 + col] = pk;
    }
    // Stage B tile: W rows n0..n0+63 (out = x @ W^T => B[k][n] = W[n][k])
#pragma unroll
    for (int i = 0; i < 4; ++i) {
      int p = tid + i * 256;
      int row = p >> 4, col = (p & 15) * 2;
      float2 f = *(const float2*)&W[(size_t)(n0 + row) * DM + k0 + col];
      unsigned int pk = (unsigned int)f2bf(f.x) | ((unsigned int)f2bf(f.y) << 16);
      *(unsigned int*)&B_lds[row * 34 + col] = pk;
    }
    if (k0 + 32 < DM)  // hint: prefetch next A tile rows for this wave
      __builtin_prefetch(&x[(size_t)(m0 + wave * 16 + l16) * DM + k0 + 32], 0, 3);
    __syncthreads();

    Frag a;
#pragma unroll
    for (int j = 0; j < 8; ++j)
      a.u[j] = *(const unsigned int*)&A_lds[(wave * 16 + l16) * 34 + ka_off(j, hi)];
#pragma unroll
    for (int t = 0; t < 4; ++t) {
      Frag b;
#pragma unroll
      for (int j = 0; j < 8; ++j)
        b.u[j] = *(const unsigned int*)&B_lds[(t * 16 + l16) * 34 + kb_off(j, hi)];
      acc[t].v = __builtin_amdgcn_wmma_f32_16x16x32_bf16(
          false, a.v, false, b.v, (short)0, acc[t].v, false, false);
    }
  }

  const bool do_rope = (z < 2);
#pragma unroll
  for (int t = 0; t < 4; ++t) {
    int n = n0 + t * 16 + l16;
    int h = n / HD, d = n & (HD - 1);
    float invf = __powf(10000.0f, -(float)(d >> 1) * (1.0f / 32.0f));
#pragma unroll
    for (int r = 0; r < 8; ++r) {
      int m = m0 + wave * 16 + r + 8 * hi;
      int b = m >> 11, s = m & (S_LEN - 1);
      float vv = acc[t].f[r];
      float pr = __shfl_xor(vv, 1, 32);   // partner column (even<->odd)
      float res = vv;
      if (do_rope) {
        float sn, cs;
        __sincosf((float)s * invf, &sn, &cs);
        res = (d & 1) ? (pr * sn + vv * cs) : (vv * cs - pr * sn);
      }
      outp[(((size_t)(b * N_HEAD + h)) * S_LEN + s) * HD + d] = f2bf(res);
    }
  }
}

// ---------------------------------------------------------------------------
// Kernel 2: causal attention with erf(score/8)+1 weighting, flash-style.
// One WG = 128 query rows of one (b,h); wave owns 16 rows.
// ---------------------------------------------------------------------------
__global__ __launch_bounds__(256)
void attn_kernel(const unsigned short* __restrict__ q,
                 const unsigned short* __restrict__ k,
                 const unsigned short* __restrict__ v,
                 unsigned short* __restrict__ y) {
  __shared__ unsigned short K_lds[64 * 66];        // [t][d], d-contiguous
  __shared__ unsigned short Vt_lds[64 * 66];       // [d][t], t-contiguous
  __shared__ unsigned short P_lds[8 * 16 * 66];    // per-wave [row][t]

  const int tid = threadIdx.x;
  const int wave = tid >> 5, lane = tid & 31, l16 = lane & 15, hi = lane >> 4;
  const int bh = blockIdx.y;
  const int qm0 = blockIdx.x * 128;
  const int qmw = qm0 + wave * 16;
  const size_t base = (size_t)bh * S_LEN * HD;

  // Q A-fragments for K=0..31 and 32..63 (bf16 head-major, read direct)
  Frag qf[2];
#pragma unroll
  for (int kk = 0; kk < 2; ++kk)
#pragma unroll
    for (int j = 0; j < 8; ++j)
      qf[kk].u[j] = *(const unsigned int*)&q[base + (size_t)(qmw + l16) * HD +
                                             kk * 32 + ka_off(j, hi)];

  Acc yacc[4];
  float rs[8];
#pragma unroll
  for (int t = 0; t < 4; ++t)
#pragma unroll
    for (int r = 0; r < 8; ++r) yacc[t].f[r] = 0.0f;
#pragma unroll
  for (int r = 0; r < 8; ++r) rs[r] = 0.0f;

  unsigned short* myP = &P_lds[wave * 16 * 66];
  const int nkb = 2 * blockIdx.x + 2;

  for (int kb = 0; kb < nkb; ++kb) {
    const int t0 = kb * 64;
    __syncthreads();
    // Stage K [t][d] and V transposed [d][t]
#pragma unroll
    for (int i = 0; i < 8; ++i) {
      int p = tid + i * 256;
      int t = p >> 5, dc = (p & 31) * 2;
      unsigned int kw = *(const unsigned int*)&k[base + (size_t)(t0 + t) * HD + dc];
      *(unsigned int*)&K_lds[t * 66 + dc] = kw;
      unsigned int vw = *(const unsigned int*)&v[base + (size_t)(t0 + t) * HD + dc];
      Vt_lds[(dc)     * 66 + t] = (unsigned short)(vw & 0xffffu);
      Vt_lds[(dc + 1) * 66 + t] = (unsigned short)(vw >> 16);
    }
    __syncthreads();

    if (t0 <= qmw + 15) {                 // wave-uniform causal block skip
      // S = Q @ K^T   (B[k=d][n=t] = K_lds[t][d])
      Acc sacc[4];
#pragma unroll
      for (int a = 0; a < 4; ++a) {
#pragma unroll
        for (int r = 0; r < 8; ++r) sacc[a].f[r] = 0.0f;
#pragma unroll
        for (int kk = 0; kk < 2; ++kk) {
          Frag b;
#pragma unroll
          for (int j = 0; j < 8; ++j)
            b.u[j] = *(const unsigned int*)&K_lds[(a * 16 + l16) * 66 +
                                                  kk * 32 + kb_off(j, hi)];
          sacc[a].v = __builtin_amdgcn_wmma_f32_16x16x32_bf16(
              false, qf[kk].v, false, b.v, (short)0, sacc[a].v, false, false);
        }
      }
      // weights = erf(s/8)+1, causal mask, rowsum, stage P for re-fragmenting
#pragma unroll
      for (int a = 0; a < 4; ++a) {
        int tg = t0 + a * 16 + l16;
#pragma unroll
        for (int r = 0; r < 8; ++r) {
          int sg = qmw + r + 8 * hi;
          float w = erff(sacc[a].f[r] * 0.125f) + 1.0f;
          if (tg > sg) w = 0.0f;
          rs[r] += w;
          myP[(r + 8 * hi) * 66 + a * 16 + l16] = f2bf(w);
        }
      }
      // Y += P @ V   (A from myP, B[k=t][n=d] = Vt_lds[d][t])
#pragma unroll
      for (int kk = 0; kk < 2; ++kk) {
        Frag pa;
#pragma unroll
        for (int j = 0; j < 8; ++j)
          pa.u[j] = *(const unsigned int*)&myP[l16 * 66 + kk * 32 + ka_off(j, hi)];
#pragma unroll
        for (int ad = 0; ad < 4; ++ad) {
          Frag b;
#pragma unroll
          for (int j = 0; j < 8; ++j)
            b.u[j] = *(const unsigned int*)&Vt_lds[(ad * 16 + l16) * 66 +
                                                   kk * 32 + kb_off(j, hi)];
          yacc[ad].v = __builtin_amdgcn_wmma_f32_16x16x32_bf16(
              false, pa.v, false, b.v, (short)0, yacc[ad].v, false, false);
        }
      }
    }
  }

  // row-sum butterfly over the 16-lane group holding each row
#pragma unroll
  for (int r = 0; r < 8; ++r) {
    float t = rs[r];
    t += __shfl_xor(t, 1, 32);
    t += __shfl_xor(t, 2, 32);
    t += __shfl_xor(t, 4, 32);
    t += __shfl_xor(t, 8, 32);
    rs[r] = fmaxf(t, 1e-6f);
  }
  const int b = bh / N_HEAD, h = bh % N_HEAD;
#pragma unroll
  for (int ad = 0; ad < 4; ++ad) {
    int d = ad * 16 + l16;
#pragma unroll
    for (int r = 0; r < 8; ++r) {
      int s = qmw + r + 8 * hi;
      size_t m = (size_t)b * S_LEN + s;
      y[m * DM + h * HD + d] = f2bf(yacc[ad].f[r] / rs[r]);
    }
  }
}

// ---------------------------------------------------------------------------
// Kernel 3: out = y @ Wproj^T, bf16 A from workspace, f32 output.
// ---------------------------------------------------------------------------
__global__ __launch_bounds__(256)
void proj_kernel(const unsigned short* __restrict__ y,
                 const float* __restrict__ Wp,
                 float* __restrict__ out) {
  __shared__ unsigned short A_lds[128 * 34];
  __shared__ unsigned short B_lds[64 * 34];

  const int tid = threadIdx.x;
  const int wave = tid >> 5, lane = tid & 31, l16 = lane & 15, hi = lane >> 4;
  const int m0 = blockIdx.x * 128;
  const int n0 = blockIdx.y * 64;

  Acc acc[4];
#pragma unroll
  for (int t = 0; t < 4; ++t)
#pragma unroll
    for (int r = 0; r < 8; ++r) acc[t].f[r] = 0.0f;

  for (int k0 = 0; k0 < DM; k0 += 32) {
    __syncthreads();
#pragma unroll
    for (int i = 0; i < 8; ++i) {
      int p = tid + i * 256;
      int row = p >> 4, col = (p & 15) * 2;
      *(unsigned int*)&A_lds[row * 34 + col] =
          *(const unsigned int*)&y[(size_t)(m0 + row) * DM + k0 + col];
    }
#pragma unroll
    for (int i = 0; i < 4; ++i) {
      int p = tid + i * 256;
      int row = p >> 4, col = (p & 15) * 2;
      float2 f = *(const float2*)&Wp[(size_t)(n0 + row) * DM + k0 + col];
      unsigned int pk = (unsigned int)f2bf(f.x) | ((unsigned int)f2bf(f.y) << 16);
      *(unsigned int*)&B_lds[row * 34 + col] = pk;
    }
    __syncthreads();

    Frag a;
#pragma unroll
    for (int j = 0; j < 8; ++j)
      a.u[j] = *(const unsigned int*)&A_lds[(wave * 16 + l16) * 34 + ka_off(j, hi)];
#pragma unroll
    for (int t = 0; t < 4; ++t) {
      Frag b;
#pragma unroll
      for (int j = 0; j < 8; ++j)
        b.u[j] = *(const unsigned int*)&B_lds[(t * 16 + l16) * 34 + kb_off(j, hi)];
      acc[t].v = __builtin_amdgcn_wmma_f32_16x16x32_bf16(
          false, a.v, false, b.v, (short)0, acc[t].v, false, false);
    }
  }

#pragma unroll
  for (int t = 0; t < 4; ++t) {
    int n = n0 + t * 16 + l16;
#pragma unroll
    for (int r = 0; r < 8; ++r) {
      int m = m0 + wave * 16 + r + 8 * hi;
      out[(size_t)m * DM + n] = acc[t].f[r];
    }
  }
}

// ---------------------------------------------------------------------------
extern "C" void kernel_launch(void* const* d_in, const int* in_sizes, int n_in,
                              void* d_out, int out_size, void* d_ws, size_t ws_size,
                              hipStream_t stream) {
  const float* x  = (const float*)d_in[0];
  const float* Wq = (const float*)d_in[1];
  const float* Wk = (const float*)d_in[2];
  const float* Wv = (const float*)d_in[3];
  const float* Wp = (const float*)d_in[4];
  float* out = (float*)d_out;

  const size_t HE = (size_t)2 * N_HEAD * S_LEN * HD;  // 3,145,728 elems
  unsigned short* qws = (unsigned short*)d_ws;
  unsigned short* kws = qws + HE;
  unsigned short* vws = kws + HE;
  unsigned short* yws = vws + HE;   // [4096 x 768] bf16; total ws = 24 MB

  dim3 blk(256);
  qkv_rope_kernel<<<dim3(32, 12, 3), blk, 0, stream>>>(x, Wq, Wk, Wv, qws, kws, vws);
  attn_kernel<<<dim3(16, 24), blk, 0, stream>>>(qws, kws, vws, yws);
  proj_kernel<<<dim3(32, 12), blk, 0, stream>>>(yws, Wp, out);
}
